// EncoderLayer_87342454931889
// MI455X (gfx1250) — compile-verified, matrix operand order checked by reading, not run on hardware
//
#include <hip/hip_runtime.h>
#include <hip/hip_bf16.h>
#include <math.h>

// ---------------------------------------------------------------------------
// EncoderLayer for MI455X (gfx1250)
//   - bf16 WMMA GEMMs with pre-converted operands (b128 fragment loads,
//     no in-loop conversion VALU)
//   - atomic sparse segment-softmax for the two bipartite attentions
// ---------------------------------------------------------------------------

#define HID    256
#define HEADS  8
#define DHEAD  32
#define N_T    32768
#define N_C    32768
#define N_ALL  (N_T + N_C)
#define SCALEF 0.17677669529663687f   // 32^-0.5

typedef __attribute__((ext_vector_type(16))) __bf16 bf16x16;
typedef __attribute__((ext_vector_type(8)))  __bf16 bf16x8;
typedef __attribute__((ext_vector_type(8)))  float  f32x8;

__device__ __forceinline__ __bf16 f2bf(float f) {
  unsigned u = __builtin_bit_cast(unsigned, f);
  u += 0x7FFFu + ((u >> 16) & 1u);        // round-to-nearest-even
  unsigned short h = (unsigned short)(u >> 16);
  return __builtin_bit_cast(__bf16, h);
}

// ------------------------------ LayerNorm ----------------------------------
// one wave32 per row of 256; each lane owns 8 contiguous floats; bf16 output
__global__ __launch_bounds__(256) void ln_bf_kernel(
    const float* __restrict__ x, const float* __restrict__ g,
    const float* __restrict__ b, __bf16* __restrict__ y, int nrows)
{
  int wid  = (int)((blockIdx.x * blockDim.x + threadIdx.x) >> 5);
  int lane = threadIdx.x & 31;
  if (wid >= nrows) return;
  const float* xr = x + (size_t)wid * HID;
  float v[8];
  float s = 0.f;
#pragma unroll
  for (int i = 0; i < 8; ++i) { v[i] = xr[lane * 8 + i]; s += v[i]; }
#pragma unroll
  for (int o = 16; o > 0; o >>= 1) s += __shfl_xor(s, o, 32);
  float mu = s * (1.0f / HID);
  float q = 0.f;
#pragma unroll
  for (int i = 0; i < 8; ++i) { float d = v[i] - mu; q += d * d; }
#pragma unroll
  for (int o = 16; o > 0; o >>= 1) q += __shfl_xor(q, o, 32);
  float inv = rsqrtf(q * (1.0f / HID) + 1e-5f);
  __bf16* yr = y + (size_t)wid * HID + lane * 8;
#pragma unroll
  for (int i = 0; i < 8; ++i) {
    int n = lane * 8 + i;
    yr[i] = f2bf((v[i] - mu) * inv * g[n] + b[n]);
  }
}

// ---------------- weight transpose + convert: W(KxN) -> Wt(NxK) bf16 -------
__global__ __launch_bounds__(256) void wtrans_kernel(
    const float* __restrict__ W, __bf16* __restrict__ Wt)
{
  int t = blockIdx.x * blockDim.x + threadIdx.x;   // 65536 threads
  int n = t >> 8, k = t & 255;
  Wt[(size_t)n * HID + k] = f2bf(W[(size_t)k * HID + n]);
}

// ---------------------- f32 -> bf16 bulk conversion ------------------------
__global__ __launch_bounds__(256) void cvt_bf16_kernel(
    const float* __restrict__ src, __bf16* __restrict__ dst, int n8)
{
  int i = blockIdx.x * blockDim.x + threadIdx.x;   // one thread per 8 elems
  if (i >= n8) return;
  const float4* sp = (const float4*)(src + (size_t)i * 8);
  float4 a = sp[0], b = sp[1];
  __bf16* dp = dst + (size_t)i * 8;
  dp[0] = f2bf(a.x); dp[1] = f2bf(a.y); dp[2] = f2bf(a.z); dp[3] = f2bf(a.w);
  dp[4] = f2bf(b.x); dp[5] = f2bf(b.y); dp[6] = f2bf(b.z); dp[7] = f2bf(b.w);
}

// ------------------------------ WMMA GEMM ----------------------------------
// C[M x 256] = epilogue(A[M x 256] @ W[256 x 256] + bias [+resid])
// A bf16 row-major; Bt bf16 N-major (Bt[n*256+k] = W[k][n]).
// One wave computes a 32x64 strip: 2 M-tiles x 4 N-tiles, 8 WMMAs / K-step.
// Fragment loads are two contiguous b128 loads per 16x32 tile per lane
// (ISA 7.12.2: lane half h owns K = h*8..h*8+7 and 16+h*8..16+h*8+7).
// mode: 0 = f32 out, 1 = f32 out + resid, 2 = bf16 out with exact GELU
__global__ __launch_bounds__(256) void gemm256_kernel(
    const __bf16* __restrict__ A, const __bf16* __restrict__ Bt,
    const float* __restrict__ bias, const float* __restrict__ resid,
    float* __restrict__ outF, __bf16* __restrict__ outH, int M, int mode)
{
  int wid   = (int)((blockIdx.x * blockDim.x + threadIdx.x) >> 5);
  int lane  = threadIdx.x & 31;
  int mblk  = wid >> 2;       // 32-row block
  int nquad = wid & 3;        // 64-col quad
  if (mblk * 32 >= M) return; // wave-uniform; EXEC stays all-ones for WMMA
  int half = lane >> 4;
  int lrow = lane & 15;
  int hk   = half * 8;

  f32x8 acc[2][4] = {};
  for (int kb = 0; kb < HID; kb += 32) {
    bf16x16 afrag[2];
#pragma unroll
    for (int mi = 0; mi < 2; ++mi) {
      const __bf16* ap = A + (size_t)(mblk * 32 + mi * 16 + lrow) * HID + kb + hk;
      bf16x8 lo = *(const bf16x8*)(ap);
      bf16x8 hi = *(const bf16x8*)(ap + 16);
      afrag[mi] = __builtin_shufflevector(lo, hi, 0, 1, 2, 3, 4, 5, 6, 7,
                                          8, 9, 10, 11, 12, 13, 14, 15);
    }
#pragma unroll
    for (int j = 0; j < 4; ++j) {
      int n = (nquad * 4 + j) * 16 + lrow;
      const __bf16* bp = Bt + (size_t)n * HID + kb + hk;
      bf16x8 lo = *(const bf16x8*)(bp);
      bf16x8 hi = *(const bf16x8*)(bp + 16);
      bf16x16 bfrag = __builtin_shufflevector(lo, hi, 0, 1, 2, 3, 4, 5, 6, 7,
                                              8, 9, 10, 11, 12, 13, 14, 15);
#pragma unroll
      for (int mi = 0; mi < 2; ++mi)
        acc[mi][j] = __builtin_amdgcn_wmma_f32_16x16x32_bf16(
            false, afrag[mi], false, bfrag, (short)0, acc[mi][j], false, false);
    }
  }
  // epilogue: C/D layout VGPR r -> M = r + 8*half, N = lane%16
#pragma unroll
  for (int mi = 0; mi < 2; ++mi) {
#pragma unroll
    for (int j = 0; j < 4; ++j) {
      int n = (nquad * 4 + j) * 16 + lrow;
#pragma unroll
      for (int r = 0; r < 8; ++r) {
        int mm = mblk * 32 + mi * 16 + r + 8 * half;
        float v = acc[mi][j][r] + bias[n];
        if (mode == 1) {
          v += resid[(size_t)mm * HID + n];
          outF[(size_t)mm * HID + n] = v;
        } else if (mode == 2) {
          v = 0.5f * v * (1.0f + erff(v * 0.70710678118654752f));
          outH[(size_t)mm * HID + n] = f2bf(v);
        } else {
          outF[(size_t)mm * HID + n] = v;
        }
      }
    }
  }
}

// ------------------------------ fill ---------------------------------------
__global__ void fill_kernel(float* __restrict__ p, float val, int n) {
  int i = blockIdx.x * blockDim.x + threadIdx.x;
  if (i < n) p[i] = val;
}

// --------------------------- edge pass 1: scores ---------------------------
__global__ __launch_bounds__(256) void edge_score_kernel(
    const float* __restrict__ Q, const float* __restrict__ K,
    const float* __restrict__ bias, const int* __restrict__ row,
    const int* __restrict__ col, int qoff, int kvoff,
    float* __restrict__ sbuf, float* __restrict__ mbuf, int E)
{
  int t = blockIdx.x * blockDim.x + threadIdx.x;
  if (t >= E * HEADS) return;
  int e = t >> 3, h = t & 7;
  int r = row[e], c = col[e];
  const float4* qp = (const float4*)(Q + (size_t)(qoff + r) * HID + h * DHEAD);
  const float4* kp = (const float4*)(K + (size_t)(kvoff + c) * HID + h * DHEAD);
  float s = 0.f;
#pragma unroll
  for (int i = 0; i < 8; ++i) {
    float4 a = qp[i], bb = kp[i];
    s += a.x * bb.x + a.y * bb.y + a.z * bb.z + a.w * bb.w;
  }
  s = s * SCALEF + bias[t];
  sbuf[t] = s;
  float* mp = mbuf + (size_t)r * HEADS + h;
  if (s >= 0.f) atomicMax((int*)mp, __float_as_int(s));           // pos: signed max
  else          atomicMin((unsigned int*)mp, __float_as_uint(s)); // neg: unsigned min
}

// --------------------------- edge pass 2: exp/sum --------------------------
__global__ __launch_bounds__(256) void edge_exp_kernel(
    float* __restrict__ sbuf, const float* __restrict__ mbuf,
    const int* __restrict__ row, float* __restrict__ zbuf, int E)
{
  int t = blockIdx.x * blockDim.x + threadIdx.x;
  if (t >= E * HEADS) return;
  int e = t >> 3, h = t & 7;
  int r = row[e];
  float ee = expf(sbuf[t] - mbuf[(size_t)r * HEADS + h]);
  sbuf[t] = ee;
  atomicAdd(zbuf + (size_t)r * HEADS + h, ee);
}

// ------------------------ edge pass 3: weighted V --------------------------
__global__ __launch_bounds__(256) void edge_accum_kernel(
    const float* __restrict__ sbuf, const float* __restrict__ zbuf,
    const float* __restrict__ V, const int* __restrict__ row,
    const int* __restrict__ col, int kvoff, int outoff,
    float* __restrict__ accum, int E)
{
  int t = blockIdx.x * blockDim.x + threadIdx.x;
  if (t >= E * HEADS) return;
  int e = t >> 3, h = t & 7;
  int r = row[e], c = col[e];
  float a = sbuf[t] / (zbuf[(size_t)r * HEADS + h] + 1e-16f);
  const float* vp = V + (size_t)(kvoff + c) * HID + h * DHEAD;
  float* op = accum + (size_t)(outoff + r) * HID + h * DHEAD;
#pragma unroll
  for (int i = 0; i < DHEAD; ++i) atomicAdd(op + i, a * vp[i]);
}

// ---------------------------------------------------------------------------
extern "C" void kernel_launch(void* const* d_in, const int* in_sizes, int n_in,
                              void* d_out, int out_size, void* d_ws, size_t ws_size,
                              hipStream_t stream) {
  const float* x_all    = (const float*)d_in[0];
  const float* bias_c2t = (const float*)d_in[1];
  const float* bias_t2c = (const float*)d_in[2];
  const int*   t2c_row  = (const int*)d_in[3];
  const int*   t2c_col  = (const int*)d_in[4];
  const int*   c2t_row  = (const int*)d_in[5];
  const int*   c2t_col  = (const int*)d_in[6];
  const float* ln1_g = (const float*)d_in[7];
  const float* ln1_b = (const float*)d_in[8];
  const float* Wq = (const float*)d_in[9];   const float* bq = (const float*)d_in[10];
  const float* Wk = (const float*)d_in[11];  const float* bk = (const float*)d_in[12];
  const float* Wv = (const float*)d_in[13];  const float* bv = (const float*)d_in[14];
  const float* Wo = (const float*)d_in[15];  const float* bo = (const float*)d_in[16];
  const float* ln2_g = (const float*)d_in[17];
  const float* ln2_b = (const float*)d_in[18];
  const float* W1 = (const float*)d_in[19];  const float* b1 = (const float*)d_in[20];
  const float* W2 = (const float*)d_in[21];  const float* b2 = (const float*)d_in[22];

  const int E = in_sizes[3];                       // 1,000,000
  const size_t MATE = (size_t)N_ALL * HID;         // 16.78M elements
  const size_t WEL  = (size_t)HID * HID;           // 65536 elements

  // -------- workspace layout (bytes) --------
  char* w = (char*)d_ws;
  __bf16* y1bf  = (__bf16*)w;           w += MATE * 2;   // LN out / attn bf16 / LN2 out
  __bf16* WtQ   = (__bf16*)w;           w += WEL * 2;
  __bf16* WtK   = (__bf16*)w;           w += WEL * 2;
  __bf16* WtV   = (__bf16*)w;           w += WEL * 2;
  __bf16* WtO   = (__bf16*)w;           w += WEL * 2;
  __bf16* Wt1   = (__bf16*)w;           w += WEL * 2;
  __bf16* Wt2   = (__bf16*)w;           w += WEL * 2;
  float*  QA    = (float*)w;            w += MATE * 4;   // Q (later: FFN1 bf16 out)
  float*  KA    = (float*)w;            w += MATE * 4;   // K
  float*  VA    = (float*)w;            w += MATE * 4;   // V (later: x + attn resid)
  float*  attn  = (float*)w;            w += MATE * 4;   // concat [y_target_, y_context_]
  float*  sbuf  = (float*)w;            w += (size_t)E * HEADS * 4;
  float*  mbuf  = (float*)w;            w += (size_t)N_T * HEADS * 4;
  float*  zbuf  = (float*)w;            w += (size_t)N_T * HEADS * 4;
  float*  xres  = VA;                   // alias: V dead after attention
  __bf16* ff1bf = (__bf16*)QA;          // alias: Q dead after attention

  const int gemmBlocks = (N_ALL / 32) * 4 * 32 / 256;  // 1024
  const int eblocks    = (E * HEADS + 255) / 256;
  const int mzblocks   = (N_T * HEADS + 255) / 256;
  const int wtBlocks   = (int)(WEL / 256);
  const int cvtBlocks  = (int)(MATE / 8 / 256);

  // 0) weights -> bf16, N-major (tiny, 6 x 64K elems)
  wtrans_kernel<<<wtBlocks, 256, 0, stream>>>(Wq, WtQ);
  wtrans_kernel<<<wtBlocks, 256, 0, stream>>>(Wk, WtK);
  wtrans_kernel<<<wtBlocks, 256, 0, stream>>>(Wv, WtV);
  wtrans_kernel<<<wtBlocks, 256, 0, stream>>>(Wo, WtO);
  wtrans_kernel<<<wtBlocks, 256, 0, stream>>>(W1, Wt1);
  wtrans_kernel<<<wtBlocks, 256, 0, stream>>>(W2, Wt2);

  // 1) LN1 over all 65536 rows -> bf16
  ln_bf_kernel<<<N_ALL / 8, 256, 0, stream>>>(x_all, ln1_g, ln1_b, y1bf, N_ALL);

  // 2) Q/K/V projections for all rows (bf16 WMMA, f32 accumulate -> f32)
  gemm256_kernel<<<gemmBlocks, 256, 0, stream>>>(y1bf, WtQ, bq, nullptr, QA, nullptr, N_ALL, 0);
  gemm256_kernel<<<gemmBlocks, 256, 0, stream>>>(y1bf, WtK, bk, nullptr, KA, nullptr, N_ALL, 0);
  gemm256_kernel<<<gemmBlocks, 256, 0, stream>>>(y1bf, WtV, bv, nullptr, VA, nullptr, N_ALL, 0);

  hipMemsetAsync(attn, 0, MATE * sizeof(float), stream);

  // 3a) attention tgt2cxt: q = context rows (offset N_T), k/v = target rows
  //     -> output rows [0, N_C)
  fill_kernel<<<mzblocks, 256, 0, stream>>>(mbuf, -INFINITY, N_T * HEADS);
  hipMemsetAsync(zbuf, 0, (size_t)N_T * HEADS * sizeof(float), stream);
  edge_score_kernel<<<eblocks, 256, 0, stream>>>(QA, KA, bias_t2c, t2c_row, t2c_col,
                                                 N_T, 0, sbuf, mbuf, E);
  edge_exp_kernel<<<eblocks, 256, 0, stream>>>(sbuf, mbuf, t2c_row, zbuf, E);
  edge_accum_kernel<<<eblocks, 256, 0, stream>>>(sbuf, zbuf, VA, t2c_row, t2c_col,
                                                 0, 0, attn, E);

  // 3b) attention cxt2tgt: q = target rows (offset 0), k/v = context rows
  //     -> output rows [N_C, N_ALL)
  fill_kernel<<<mzblocks, 256, 0, stream>>>(mbuf, -INFINITY, N_T * HEADS);
  hipMemsetAsync(zbuf, 0, (size_t)N_T * HEADS * sizeof(float), stream);
  edge_score_kernel<<<eblocks, 256, 0, stream>>>(QA, KA, bias_c2t, c2t_row, c2t_col,
                                                 0, N_T, sbuf, mbuf, E);
  edge_exp_kernel<<<eblocks, 256, 0, stream>>>(sbuf, mbuf, c2t_row, zbuf, E);
  edge_accum_kernel<<<eblocks, 256, 0, stream>>>(sbuf, zbuf, VA, c2t_row, c2t_col,
                                                 N_T, N_C, attn, E);

  // 4) attn -> bf16, then output projection + residual: xres = x_all + attn@Wo + bo
  cvt_bf16_kernel<<<cvtBlocks, 256, 0, stream>>>(attn, y1bf, (int)(MATE / 8));
  gemm256_kernel<<<gemmBlocks, 256, 0, stream>>>(y1bf, WtO, bo, x_all, xres, nullptr, N_ALL, 1);

  // 5) LN2 -> bf16
  ln_bf_kernel<<<N_ALL / 8, 256, 0, stream>>>(xres, ln2_g, ln2_b, y1bf, N_ALL);

  // 6) FFN: gelu(y1 @ W1 + b1) [bf16] @ W2 + b2 + xres -> d_out
  gemm256_kernel<<<gemmBlocks, 256, 0, stream>>>(y1bf, Wt1, b1, nullptr, nullptr, ff1bf, N_ALL, 2);
  gemm256_kernel<<<gemmBlocks, 256, 0, stream>>>(ff1bf, Wt2, b2, xres, (float*)d_out, nullptr, N_ALL, 1);
}
